// Local2DAttention_79207786873330
// MI455X (gfx1250) — matrix operand order repaired and checked by reading, not compiled, not run on hardware
//
#include <hip/hip_runtime.h>
#include <hip/hip_bf16.h>

// ---------------------------------------------------------------------------
// Local 2D window attention (with the reference's reshape quirk) for gfx1250.
// Pipeline: pack(Y,W1,W2) -> GEMM1(QKV, bf16 WMMA, double-buffered) ->
// fused attention per (t,head) block (bf16 WMMA + wave32 softmax) ->
// GEMM3(out proj, bf16 WMMA, double-buffered).
// ---------------------------------------------------------------------------

typedef __bf16 bf16_t;
typedef __attribute__((ext_vector_type(8)))  __bf16 v8bf;
typedef __attribute__((ext_vector_type(16))) __bf16 v16bf;
typedef __attribute__((ext_vector_type(8)))  float  v8f;

#define E_DIM 512
#define NTOK  16384          // M * WW = 256 * 64 tokens
#define SEQ   256            // attention sequence length (windows)
#define WW    64             // attention batch (intra-window positions)
#define NHEAD 8
#define HD    64

// ---- WMMA wrapper ---------------------------------------------------------
__device__ __forceinline__ v8f wmma_bf16(v16bf a, v16bf b, v8f c) {
  return __builtin_amdgcn_wmma_f32_16x16x32_bf16(false, a, false, b,
                                                 (short)0, c, false, false);
}

// ---- Fragment loaders (ISA 7.12.2 layouts, wave32) ------------------------
// A (16x32 bf16): lane L holds row m=L%16; element j -> k = j + 8*half + 8*(j/8)
// => two contiguous 8-element runs at k0+8*half and k0+16+8*half.
__device__ __forceinline__ v16bf frag_a(const bf16_t* p, int ld, int row0, int k0) {
  int lane = threadIdx.x & 31;
  int m    = row0 + (lane & 15);
  int half = lane >> 4;
  const bf16_t* r = p + m * ld + k0 + half * 8;
  v8bf lo = *(const v8bf*)r;
  v8bf hi = *(const v8bf*)(r + 16);
  return __builtin_shufflevector(lo, hi, 0,1,2,3,4,5,6,7,8,9,10,11,12,13,14,15);
}

// B (32x16 bf16): lane L holds column n=L%16; element j -> k = k0 + j + 8*half + 8*(j/8).
// Source is (N,K) row-major => two contiguous 8-element runs.
__device__ __forceinline__ v16bf frag_b(const bf16_t* p, int ld, int n0, int k0) {
  int lane = threadIdx.x & 31;
  int n    = n0 + (lane & 15);
  int half = lane >> 4;
  const bf16_t* r = p + n * ld + k0 + half * 8;
  v8bf lo = *(const v8bf*)r;
  v8bf hi = *(const v8bf*)(r + 16);
  return __builtin_shufflevector(lo, hi, 0,1,2,3,4,5,6,7,8,9,10,11,12,13,14,15);
}

// ---- Pack kernels ---------------------------------------------------------
__global__ __launch_bounds__(256) void k_pack_bf16(const float* __restrict__ s,
                                                   bf16_t* __restrict__ d, int n) {
  int i = blockIdx.x * 256 + threadIdx.x;
  if (i < n) d[i] = (bf16_t)s[i];
}

// Y[r][e2] = x[b, pos, ch] per the reference's reshape bijection.
__global__ __launch_bounds__(256) void k_pack_y(const float* __restrict__ x,
                                                bf16_t* __restrict__ yb) {
  int id = blockIdx.x * 256 + threadIdx.x;       // total = NTOK*E = 8388608
  int r  = id >> 9;          // token row
  int e2 = id & 511;
  int m  = r >> 6;           // window index
  int t  = r & 63;           // intra-window token
  int b  = m >> 4;
  int hy = (m >> 2) & 3;
  int hx = m & 3;
  int ch = t * 8 + (e2 >> 6);
  int rr = e2 & 63;
  int iy = rr >> 3, ix = rr & 7;
  int pos = (hy * 8 + iy) * 32 + hx * 8 + ix;
  yb[id] = (bf16_t)x[((size_t)b * 1024 + pos) * E_DIM + ch];
}

// ---------------------------------------------------------------------------
// Shared double-buffered bf16 GEMM mainloop: C(256x128 block) = A @ B^T,
// A = (rows, 512) row-major bf16, B = (cols, 512) row-major bf16 (i.e. W as
// stored). 8 waves in a 4(m) x 2(n) grid; each wave owns a 64x64 tile
// (4x4 WMMA accumulators). One barrier per K-step; global loads for step
// i+1 are register-staged while step i's WMMAs run from the other buffer.
// acc is a v8f[4][4] declared and zero-initialized by the macro.
// ---------------------------------------------------------------------------
#define GEMM_MAINLOOP(Aptr, Bptr)                                              \
  __shared__ bf16_t AsBuf[2][256 * 40];                                        \
  __shared__ bf16_t BsBuf[2][128 * 40];                                        \
  int tid = threadIdx.x, lane = tid & 31, w = tid >> 5;                        \
  int wm = w >> 1, wn = w & 1;                                                 \
  int R0 = blockIdx.x * 256, C0 = blockIdx.y * 128;                            \
  int brow = tid >> 1, bseg = tid & 1;                                         \
  v8f acc[4][4];                                                               \
  _Pragma("unroll") for (int mt = 0; mt < 4; ++mt)                             \
  _Pragma("unroll") for (int nt = 0; nt < 4; ++nt)                             \
      acc[mt][nt] = (v8f){0,0,0,0,0,0,0,0};                                    \
  v8bf aR[4], bR[2];                                                           \
  {                                                                            \
    const bf16_t* ga = (Aptr) + (size_t)(R0 + tid) * E_DIM;                    \
    _Pragma("unroll") for (int c = 0; c < 4; ++c)                              \
        aR[c] = *(const v8bf*)(ga + c * 8);                                    \
    const bf16_t* gb = (Bptr) + (size_t)(C0 + brow) * E_DIM + bseg * 16;       \
    bR[0] = *(const v8bf*)gb;                                                  \
    bR[1] = *(const v8bf*)(gb + 8);                                            \
    _Pragma("unroll") for (int c = 0; c < 4; ++c)                              \
        *(v8bf*)(&AsBuf[0][tid * 40 + c * 8]) = aR[c];                         \
    *(v8bf*)(&BsBuf[0][brow * 40 + bseg * 16])     = bR[0];                    \
    *(v8bf*)(&BsBuf[0][brow * 40 + bseg * 16 + 8]) = bR[1];                    \
  }                                                                            \
  __syncthreads();                                                             \
  for (int it = 0; it < E_DIM / 32; ++it) {                                    \
    int cur = it & 1;                                                          \
    if (it + 1 < E_DIM / 32) {                                                 \
      const bf16_t* ga = (Aptr) + (size_t)(R0 + tid) * E_DIM + (it + 1) * 32;  \
      _Pragma("unroll") for (int c = 0; c < 4; ++c)                            \
          aR[c] = *(const v8bf*)(ga + c * 8);                                  \
      const bf16_t* gb =                                                       \
          (Bptr) + (size_t)(C0 + brow) * E_DIM + (it + 1) * 32 + bseg * 16;    \
      bR[0] = *(const v8bf*)gb;                                                \
      bR[1] = *(const v8bf*)(gb + 8);                                          \
    }                                                                          \
    const bf16_t* As = AsBuf[cur];                                             \
    const bf16_t* Bs = BsBuf[cur];                                             \
    v16bf bF[4];                                                               \
    _Pragma("unroll") for (int nt = 0; nt < 4; ++nt)                           \
        bF[nt] = frag_b(Bs, 40, wn * 64 + nt * 16, 0);                         \
    _Pragma("unroll") for (int mt = 0; mt < 4; ++mt) {                         \
      v16bf aF = frag_a(As, 40, wm * 64 + mt * 16, 0);                         \
      _Pragma("unroll") for (int nt = 0; nt < 4; ++nt)                         \
          acc[mt][nt] = wmma_bf16(aF, bF[nt], acc[mt][nt]);                    \
    }                                                                          \
    if (it + 1 < E_DIM / 32) {                                                 \
      int nxt = cur ^ 1;                                                       \
      _Pragma("unroll") for (int c = 0; c < 4; ++c)                            \
          *(v8bf*)(&AsBuf[nxt][tid * 40 + c * 8]) = aR[c];                     \
      *(v8bf*)(&BsBuf[nxt][brow * 40 + bseg * 16])     = bR[0];                \
      *(v8bf*)(&BsBuf[nxt][brow * 40 + bseg * 16 + 8]) = bR[1];                \
      __syncthreads();                                                         \
    }                                                                          \
  }

// ---- GEMM1: Y(16384x512) @ W1^T(512x1536) + b -> scattered bf16 Q/K/V -----
__global__ __launch_bounds__(256) void k_gemm_qkv(
    const bf16_t* __restrict__ yb, const bf16_t* __restrict__ wb,
    const float* __restrict__ bias,
    bf16_t* __restrict__ Q, bf16_t* __restrict__ K, bf16_t* __restrict__ V) {
  GEMM_MAINLOOP(yb, wb)

  int half = lane >> 4, ln = lane & 15;
#pragma unroll
  for (int nt = 0; nt < 4; ++nt) {
    int J = C0 + wn * 64 + nt * 16 + ln;       // global output column (0..1535)
    float bv = bias[J];
    int sec  = J >> 9;                         // 0=Q 1=K 2=V
    int e    = J & 511;
    int head = e >> 6, d = e & 63;
    bf16_t* base = (sec == 0) ? Q : (sec == 1) ? K : V;
    float scale  = (sec == 0) ? 0.125f : 1.0f; // fold 1/sqrt(hd) into Q
#pragma unroll
    for (int mt = 0; mt < 4; ++mt)
#pragma unroll
      for (int i = 0; i < 8; ++i) {
        int R = R0 + wm * 64 + mt * 16 + i + half * 8;   // token row
        int m_tok = R >> 6, t = R & 63;
        size_t dst = (((size_t)(t * NHEAD + head) * SEQ + m_tok) * HD + d);
        base[dst] = (bf16_t)((acc[mt][nt][i] + bv) * scale);
      }
  }
}

// ---- GEMM3: O(16384x512) @ W2^T(512x512) + b -> un-windowed fp32 out ------
__global__ __launch_bounds__(256) void k_gemm_out(
    const bf16_t* __restrict__ Ob, const bf16_t* __restrict__ wb,
    const float* __restrict__ bias, float* __restrict__ out) {
  GEMM_MAINLOOP(Ob, wb)

  int half = lane >> 4, ln = lane & 15;
#pragma unroll
  for (int nt = 0; nt < 4; ++nt) {
    int J = C0 + wn * 64 + nt * 16 + ln;
    float bv = bias[J];
#pragma unroll
    for (int mt = 0; mt < 4; ++mt)
#pragma unroll
      for (int i = 0; i < 8; ++i) {
        int R = R0 + wm * 64 + mt * 16 + i + half * 8;
        int m = R >> 6, t = R & 63;
        int b = m >> 4, hy = (m >> 2) & 3, hx = m & 3;
        int iy = t >> 3, ix = t & 7;
        int pos = (hy * 8 + iy) * 32 + hx * 8 + ix;
        out[((size_t)b * 1024 + pos) * E_DIM + J] = acc[mt][nt][i] + bv;
      }
  }
}

// ---- Fused attention: one block per (t, head); seq=256, hd=64 -------------
__global__ __launch_bounds__(256) void k_attn(
    const bf16_t* __restrict__ Q, const bf16_t* __restrict__ K,
    const bf16_t* __restrict__ V, bf16_t* __restrict__ Ob) {
  extern __shared__ char smem[];
  bf16_t* Qs  = (bf16_t*)smem;             // [256][72]
  bf16_t* Ks  = Qs + 256 * 72;             // [256][72]
  bf16_t* Vst = Ks + 256 * 72;             // [64][264]  (V transposed: d,s)
  bf16_t* Ps  = Vst + 64 * 264;            // [256][264] (probabilities)

  int tid = threadIdx.x, lane = tid & 31, w = tid >> 5;
  int bid = blockIdx.x;                    // = t*8 + head
  int t = bid >> 3, nh = bid & 7;
  const bf16_t* Qg = Q + (size_t)bid * SEQ * HD;
  const bf16_t* Kg = K + (size_t)bid * SEQ * HD;
  const bf16_t* Vg = V + (size_t)bid * SEQ * HD;

  // Stage Q, K, V^T into LDS (coalesced 16B chunks).
#pragma unroll
  for (int c = 0; c < 8; ++c) {
    int off = (c * 256 + tid) * 8;
    int row = off >> 6, col = off & 63;
    *(v8bf*)(Qs + row * 72 + col) = *(const v8bf*)(Qg + off);
    *(v8bf*)(Ks + row * 72 + col) = *(const v8bf*)(Kg + off);
    v8bf vv = *(const v8bf*)(Vg + off);
#pragma unroll
    for (int j = 0; j < 8; ++j) Vst[(col + j) * 264 + row] = vv[j];
  }
  __syncthreads();

  // --- Scores: rows [w*32, w*32+32) vs all 256 keys ---
  int row0 = w * 32;
  int half = lane >> 4, ln = lane & 15;
  for (int st = 0; st < 16; ++st) {
    v16bf bF[2];
#pragma unroll
    for (int kc = 0; kc < 2; ++kc) bF[kc] = frag_b(Ks, 72, st * 16, kc * 32);
#pragma unroll
    for (int mt = 0; mt < 2; ++mt) {
      v8f s = (v8f){0,0,0,0,0,0,0,0};
#pragma unroll
      for (int kc = 0; kc < 2; ++kc)
        s = wmma_bf16(frag_a(Qs, 72, row0 + mt * 16, kc * 32), bF[kc], s);
#pragma unroll
      for (int i = 0; i < 8; ++i)
        Ps[(row0 + mt * 16 + i + half * 8) * 264 + st * 16 + ln] = (bf16_t)s[i];
    }
  }
  __syncthreads();

  // --- Softmax over s (row-wise); one row per wave-iteration ---
  for (int rr = 0; rr < 32; ++rr) {
    int r = row0 + rr;
    bf16_t* prow = Ps + r * 264 + lane * 8;
    v8bf pv = *(const v8bf*)prow;
    float xv[8], mx = -3.0e38f;
#pragma unroll
    for (int j = 0; j < 8; ++j) { xv[j] = (float)pv[j]; mx = fmaxf(mx, xv[j]); }
#pragma unroll
    for (int o = 16; o >= 1; o >>= 1) mx = fmaxf(mx, __shfl_xor(mx, o, 32));
    float sum = 0.f;
#pragma unroll
    for (int j = 0; j < 8; ++j) { xv[j] = __expf(xv[j] - mx); sum += xv[j]; }
#pragma unroll
    for (int o = 16; o >= 1; o >>= 1) sum += __shfl_xor(sum, o, 32);
    float inv = 1.0f / sum;
    v8bf ov;
#pragma unroll
    for (int j = 0; j < 8; ++j) ov[j] = (bf16_t)(xv[j] * inv);
    *(v8bf*)prow = ov;
  }
  __syncthreads();

  // --- O = P @ V ---
  v8f oacc[2][4];
#pragma unroll
  for (int mt = 0; mt < 2; ++mt)
#pragma unroll
    for (int dt = 0; dt < 4; ++dt) oacc[mt][dt] = (v8f){0,0,0,0,0,0,0,0};
  for (int sc = 0; sc < 8; ++sc) {
    v16bf bF[4];
#pragma unroll
    for (int dt = 0; dt < 4; ++dt) bF[dt] = frag_b(Vst, 264, dt * 16, sc * 32);
#pragma unroll
    for (int mt = 0; mt < 2; ++mt) {
      v16bf aF = frag_a(Ps, 264, row0 + mt * 16, sc * 32);
#pragma unroll
      for (int dt = 0; dt < 4; ++dt) oacc[mt][dt] = wmma_bf16(aF, bF[dt], oacc[mt][dt]);
    }
  }
  // Scatter to Ob[(l*64 + t)*512 + nh*64 + d] (token-major for out_proj).
#pragma unroll
  for (int mt = 0; mt < 2; ++mt)
#pragma unroll
    for (int dt = 0; dt < 4; ++dt)
#pragma unroll
      for (int i = 0; i < 8; ++i) {
        int l = row0 + mt * 16 + i + half * 8;
        int d = dt * 16 + ln;
        Ob[((size_t)l * WW + t) * E_DIM + nh * HD + d] = (bf16_t)oacc[mt][dt][i];
      }
}

// ---------------------------------------------------------------------------
extern "C" void kernel_launch(void* const* d_in, const int* in_sizes, int n_in,
                              void* d_out, int out_size, void* d_ws, size_t ws_size,
                              hipStream_t stream) {
  (void)in_sizes; (void)n_in; (void)out_size; (void)ws_size;
  const float* x   = (const float*)d_in[0];
  const float* w1  = (const float*)d_in[1];
  const float* b1  = (const float*)d_in[2];
  const float* w2  = (const float*)d_in[3];
  const float* b2  = (const float*)d_in[4];
  float* out = (float*)d_out;

  char* ws = (char*)d_ws;
  const size_t SZ_Y   = (size_t)NTOK * E_DIM * sizeof(bf16_t);   // 16 MB
  const size_t SZ_W1  = (size_t)3 * E_DIM * E_DIM * sizeof(bf16_t);
  const size_t SZ_W2  = (size_t)E_DIM * E_DIM * sizeof(bf16_t);
  const size_t SZ_QKV = (size_t)NTOK * E_DIM * sizeof(bf16_t);   // 16 MB each
  bf16_t* yb  = (bf16_t*)(ws);                 // also reused as O buffer
  bf16_t* w1b = (bf16_t*)(ws + SZ_Y);
  bf16_t* w2b = (bf16_t*)(ws + SZ_Y + SZ_W1);
  bf16_t* Qb  = (bf16_t*)(ws + SZ_Y + SZ_W1 + SZ_W2);
  bf16_t* Kb  = (bf16_t*)(ws + SZ_Y + SZ_W1 + SZ_W2 + SZ_QKV);
  bf16_t* Vb  = (bf16_t*)(ws + SZ_Y + SZ_W1 + SZ_W2 + 2 * SZ_QKV);
  bf16_t* Ob  = yb;  // Y is dead after GEMM1; alias for attention output

  // Pack / convert
  k_pack_y<<<(NTOK * E_DIM) / 256, 256, 0, stream>>>(x, yb);
  k_pack_bf16<<<(3 * E_DIM * E_DIM) / 256, 256, 0, stream>>>(w1, w1b, 3 * E_DIM * E_DIM);
  k_pack_bf16<<<(E_DIM * E_DIM) / 256, 256, 0, stream>>>(w2, w2b, E_DIM * E_DIM);

  // In-projection -> Q,K,V (scattered to [t][head][seq][d], Q pre-scaled)
  k_gemm_qkv<<<dim3(NTOK / 256, (3 * E_DIM) / 128), 256, 0, stream>>>(
      yb, w1b, b1, Qb, Kb, Vb);

  // Attention: 512 blocks = (t, head) pairs; 237 KB dynamic LDS per WGP.
  size_t smem = (size_t)(256 * 72 + 256 * 72 + 64 * 264 + 256 * 264) * sizeof(bf16_t);
  k_attn<<<WW * NHEAD, 256, smem, stream>>>(Qb, Kb, Vb, Ob);

  // Out-projection + inverse window scatter -> fp32 output
  k_gemm_out<<<dim3(NTOK / 256, E_DIM / 128), 256, 0, stream>>>(Ob, w2b, b2, out);
}
